// Block_17033840296551
// MI455X (gfx1250) — matrix operand order repaired and checked by reading, not compiled, not run on hardware
//
#include <hip/hip_runtime.h>

// ---------------------------------------------------------------------------
// ViG-style block for MI455X (gfx1250): all GEMMs via v_wmma_f32_16x16x32_bf16.
// Activations are kept in bf16 n-major ("transposed") layout so WMMA A/B
// fragments are contiguous 32B runs -> global_load_b128 clauses, no LDS
// staging and no f32->bf16 conversion in the hot loops.
// ---------------------------------------------------------------------------

constexpr int Bc  = 16;            // batch
constexpr int Cc  = 256;           // channels
constexpr int Nc  = 1024;          // H*W
constexpr int NBc = Bc * Nc;       // total columns
constexpr int Kk  = 9;             // knn

typedef __bf16 v16bf __attribute__((ext_vector_type(16)));
typedef float  v8f   __attribute__((ext_vector_type(8)));

union Frag { v16bf v; unsigned u[8]; };

__device__ __forceinline__ unsigned short f2bf1(float f) {
  unsigned u = __float_as_uint(f);
  u += 0x7fffu + ((u >> 16) & 1u);                 // round-to-nearest-even
  return (unsigned short)(u >> 16);
}
__device__ __forceinline__ float bf2f(unsigned short h) {
  return __uint_as_float(((unsigned)h) << 16);
}

// ---------------------------------------------------------------------------
// GEMM:  acc[o,n] = sum_c W[o,c] * X[n,c]   (X given bf16 n-major)
// Epilogue: +bias, BN_A, +res1, +res2, BN_B, relu; writes f32 [B][Cout][N]
// and/or bf16 n-major [B*N][Cout] for the next GEMM.
// block = 256 thr (8 waves), covers 32 o x 256 n; each wave: 2x2 WMMA tiles.
// ---------------------------------------------------------------------------
template<bool RELU, bool BIAS, bool BN_A, bool RES1, bool RES2, bool BN_B,
         bool WF32, bool WBT>
__global__ __launch_bounds__(256)
void gemm_bt(const unsigned* __restrict__ Wbf,  // [Cout][Cin/2] bf16 pairs
             const unsigned* __restrict__ XT,   // [B*N][Cin/2] bf16 pairs
             float*          __restrict__ Y,    // [B][Cout][N]
             unsigned short* __restrict__ YT,   // [B*N][Cout]
             const float*    __restrict__ bias, // [Cout]
             const float*    __restrict__ bnA,  // scale[Cout], shift[Cout]
             const float*    __restrict__ res1, // [B][Cout][N]
             const float*    __restrict__ res2,
             const float*    __restrict__ bnB,
             int Cin, int Cout)
{
  const int tid = threadIdx.x;
  const int wv  = tid >> 5, lane = tid & 31;
  const int hh  = lane >> 4, l16 = lane & 15;
  const int ng0 = blockIdx.x * 256;              // global row base (b*N+n)
  const int b   = ng0 / Nc;
  const int oA  = blockIdx.y * 32, oB = oA + 16;
  const int n0  = ng0 + wv * 16;                 // tile0 rows
  const int n1  = n0 + 128;                      // tile1 rows
  const int cp  = Cin >> 1;

  const unsigned* xrow0 = XT  + (size_t)(n0 + l16) * cp;
  const unsigned* xrow1 = XT  + (size_t)(n1 + l16) * cp;
  const unsigned* wrow0 = Wbf + (size_t)(oA + l16) * cp;
  const unsigned* wrow1 = Wbf + (size_t)(oB + l16) * cp;

  v8f a00 = {}, a01 = {}, a10 = {}, a11 = {};
  for (int k0 = 0; k0 < Cin; k0 += 32) {
    Frag A0, A1, B0, B1;
#pragma unroll
    for (int v = 0; v < 8; ++v) {                // A: k-pair dwords of W rows
      int kk = 16 * hh + 2 * v;
      A0.u[v] = wrow0[(k0 + kk) >> 1];
      A1.u[v] = wrow1[(k0 + kk) >> 1];
    }
#pragma unroll
    for (int v = 0; v < 8; ++v) {                // B: contiguous 32B run
      int ix = (k0 >> 1) + 8 * hh + v;
      B0.u[v] = xrow0[ix];
      B1.u[v] = xrow1[ix];
    }
    a00 = __builtin_amdgcn_wmma_f32_16x16x32_bf16(false, A0.v, false, B0.v, (short)0, a00, false, false);
    a01 = __builtin_amdgcn_wmma_f32_16x16x32_bf16(false, A0.v, false, B1.v, (short)0, a01, false, false);
    a10 = __builtin_amdgcn_wmma_f32_16x16x32_bf16(false, A1.v, false, B0.v, (short)0, a10, false, false);
    a11 = __builtin_amdgcn_wmma_f32_16x16x32_bf16(false, A1.v, false, B1.v, (short)0, a11, false, false);
  }

  auto epi = [&](int oX, int nrow, v8f acc) {
    const int ncol = nrow + l16;                 // global b*N+n
    const int nloc = ncol - b * Nc;
    unsigned short bt[8];
#pragma unroll
    for (int v = 0; v < 8; ++v) {
      int o = oX + 8 * hh + v;
      float r = acc[v];
      if (BIAS) r += bias[o];
      if (BN_A) r = r * bnA[o] + bnA[Cout + o];
      size_t offp = ((size_t)b * Cout + o) * Nc + nloc;
      if (RES1) r += res1[offp];
      if (RES2) r += res2[offp];
      if (BN_B) r = r * bnB[o] + bnB[Cout + o];
      if (RELU) r = fmaxf(r, 0.f);
      if (WF32) Y[offp] = r;
      if (WBT) bt[v] = f2bf1(r);
    }
    if (WBT) {                                   // 8 contiguous bf16 -> 16B
      uint4 pk;
      pk.x = (unsigned)bt[0] | ((unsigned)bt[1] << 16);
      pk.y = (unsigned)bt[2] | ((unsigned)bt[3] << 16);
      pk.z = (unsigned)bt[4] | ((unsigned)bt[5] << 16);
      pk.w = (unsigned)bt[6] | ((unsigned)bt[7] << 16);
      *reinterpret_cast<uint4*>(YT + (size_t)ncol * Cout + oX + 8 * hh) = pk;
    }
  };
  epi(oA, n0, a00); epi(oA, n1, a01);
  epi(oB, n0, a10); epi(oB, n1, a11);
}

// ---------------------------------------------------------------------------
// Transpose f32 (B,C,N) -> bf16 n-major (B,N,C)   (used only for input x)
// ---------------------------------------------------------------------------
__global__ __launch_bounds__(256)
void transpose_bf(const float* __restrict__ H, unsigned short* __restrict__ Ft)
{
  __shared__ float t[32][33];
  const int b = blockIdx.z, c0 = blockIdx.y * 32, n0 = blockIdx.x * 32;
  const int tx = threadIdx.x & 31, ty = threadIdx.x >> 5;
#pragma unroll
  for (int i = 0; i < 32; i += 8)
    t[ty + i][tx] = H[((size_t)b * Cc + c0 + ty + i) * Nc + n0 + tx];
  __syncthreads();
#pragma unroll
  for (int i = 0; i < 32; i += 8)
    Ft[((size_t)b * Nc + n0 + ty + i) * Cc + c0 + tx] = f2bf1(t[tx][ty + i]);
}

// ---------------------------------------------------------------------------
// L2-normalize rows of featT -> xn (bf16). 1 wave per row.
// ---------------------------------------------------------------------------
__global__ __launch_bounds__(256)
void rownorm(const unsigned short* __restrict__ Ft, unsigned short* __restrict__ Xn)
{
  const int wv = threadIdx.x >> 5, lane = threadIdx.x & 31;
  const size_t row = (size_t)blockIdx.x * 8 + wv;
  float vals[8]; float ss = 0.f;
#pragma unroll
  for (int i = 0; i < 8; ++i) {
    float f = bf2f(Ft[row * Cc + lane + 32 * i]);
    vals[i] = f; ss += f * f;
  }
#pragma unroll
  for (int off = 16; off > 0; off >>= 1) ss += __shfl_xor(ss, off, 32);
  const float inv = rsqrtf(ss + 1e-12f);
#pragma unroll
  for (int i = 0; i < 8; ++i)
    Xn[row * Cc + lane + 32 * i] = f2bf1(vals[i] * inv);
}

// ---------------------------------------------------------------------------
// sim = xn @ xn^T (bf16 WMMA), one 16-row strip in LDS, then iterative top-9.
// grid: (N/16, B). The 64MB sim tensor never touches HBM.
// ---------------------------------------------------------------------------
__global__ __launch_bounds__(256)
void sim_topk(const unsigned* __restrict__ Xn,   // [B][N][C/2] bf16 pairs
              int* __restrict__ Idx)             // [B][N][K]
{
  __shared__ float sims[16 * 1024];              // 64 KB (<= 320KB WGP LDS)
  const int tid = threadIdx.x;
  const int wv  = tid >> 5, lane = tid & 31;
  const int hh  = lane >> 4, l16 = lane & 15;
  const int b   = blockIdx.y, r0 = blockIdx.x * 16;
  const unsigned* Xb = Xn + (size_t)b * Nc * (Cc / 2);
  const int arow = (r0 + l16) * (Cc / 2);

  for (int pass = 0; pass < 8; ++pass) {
    const int c0 = pass * 128 + wv * 16;
    v8f acc = {};
    for (int k0 = 0; k0 < Cc; k0 += 32) {
      Frag A, Bm;
#pragma unroll
      for (int v = 0; v < 8; ++v) {
        int kk = 16 * hh + 2 * v;
        A.u[v]  = Xb[arow + ((k0 + kk) >> 1)];
        Bm.u[v] = Xb[(c0 + l16) * (Cc / 2) + (k0 >> 1) + 8 * hh + v];
      }
      acc = __builtin_amdgcn_wmma_f32_16x16x32_bf16(false, A.v, false, Bm.v,
                                                    (short)0, acc, false, false);
    }
#pragma unroll
    for (int v = 0; v < 8; ++v)
      sims[(8 * hh + v) * 1024 + c0 + l16] = acc[v];
  }
  __syncthreads();

  for (int rr = 0; rr < 2; ++rr) {               // wave w owns rows 2w, 2w+1
    const int r = wv * 2 + rr;
    float* row = &sims[r * 1024];
    for (int it = 0; it < Kk; ++it) {
      float best = -3.4e38f; int bi = 0;
      for (int i = 0; i < 32; ++i) {             // ascending j -> stable ties
        int j = lane + 32 * i;
        float v = row[j];
        if (v > best) { best = v; bi = j; }
      }
#pragma unroll
      for (int off = 16; off > 0; off >>= 1) {
        float ov = __shfl_xor(best, off, 32);
        int   oi = __shfl_xor(bi,   off, 32);
        if (ov > best || (ov == best && oi < bi)) { best = ov; bi = oi; }
      }
      if (lane == 0) {
        Idx[((size_t)b * Nc + r0 + r) * Kk + it] = bi;
        row[bi] = -3.4e38f;
      }
      __syncthreads();
    }
  }
}

// ---------------------------------------------------------------------------
// Edge MLP + relu + max over K.  et[b,n,o] = max_k relu(W@[xi|xj-xi]+b)
// Per k: 16x512 gathered ef tile in LDS, bf16 WMMA over all 512 outputs.
// Output written bf16 n-major for g2.
// ---------------------------------------------------------------------------
__global__ __launch_bounds__(256)
void edge_gemm(const unsigned* __restrict__ Ftp,  // [B][N][C/2] bf16 pairs
               const int*      __restrict__ Idx,  // [B][N][K]
               const unsigned* __restrict__ Wbf,  // [512][256] bf16 pairs
               const float*    __restrict__ ebias,// [512]
               unsigned short* __restrict__ ET)   // [B*N][512]
{
  __shared__ unsigned efl[16 * 256];              // 16 rows x 256 c-pairs
  const int tid = threadIdx.x;
  const int wv  = tid >> 5, lane = tid & 31;
  const int hh  = lane >> 4, l16 = lane & 15;
  const int b   = blockIdx.x >> 6;
  const int n0  = (blockIdx.x & 63) * 16;

  v8f accm[4];
#pragma unroll
  for (int ot = 0; ot < 4; ++ot)
#pragma unroll
    for (int v = 0; v < 8; ++v) accm[ot][v] = 0.f;

  for (int kk = 0; kk < Kk; ++kk) {
    __syncthreads();
#pragma unroll
    for (int e = 0; e < 16; ++e) {                // build ef tile
      int elem = tid + 256 * e;
      int r = elem >> 8, cp = elem & 255;
      int n = n0 + r;
      unsigned pv;
      if (cp < 128) {                             // xi half
        pv = Ftp[((size_t)b * Nc + n) * 128 + cp];
      } else {                                    // xj - xi half
        int j = Idx[((size_t)b * Nc + n) * Kk + kk];
        unsigned ui = Ftp[((size_t)b * Nc + n) * 128 + (cp - 128)];
        unsigned uj = Ftp[((size_t)b * Nc + j) * 128 + (cp - 128)];
        float d0 = bf2f((unsigned short)uj)         - bf2f((unsigned short)ui);
        float d1 = bf2f((unsigned short)(uj >> 16)) - bf2f((unsigned short)(ui >> 16));
        pv = (unsigned)f2bf1(d0) | ((unsigned)f2bf1(d1) << 16);
      }
      efl[r * 256 + cp] = pv;
    }
    __syncthreads();

#pragma unroll 1
    for (int ot = 0; ot < 4; ++ot) {
      const int o = ot * 128 + wv * 16 + l16;
      const unsigned* wrow = Wbf + (size_t)o * 256;
      v8f c = {};
      for (int k0 = 0; k0 < 512; k0 += 32) {
        Frag A, Bm;
#pragma unroll
        for (int v = 0; v < 8; ++v) {
          int ka = 16 * hh + 2 * v;
          A.u[v]  = efl[l16 * 256 + ((k0 + ka) >> 1)];
          Bm.u[v] = wrow[(k0 >> 1) + 8 * hh + v];
        }
        c = __builtin_amdgcn_wmma_f32_16x16x32_bf16(false, A.v, false, Bm.v,
                                                    (short)0, c, false, false);
      }
      const float bia = ebias[o];
#pragma unroll
      for (int v = 0; v < 8; ++v)
        accm[ot][v] = fmaxf(accm[ot][v], fmaxf(c[v] + bia, 0.f));
    }
  }

#pragma unroll
  for (int ot = 0; ot < 4; ++ot) {
    const int o = ot * 128 + wv * 16 + l16;
#pragma unroll
    for (int v = 0; v < 8; ++v)
      ET[((size_t)b * Nc + n0 + 8 * hh + v) * 512 + o] = f2bf1(accm[ot][v]);
  }
}

// ---------------------------------------------------------------------------
// 3x3 SAME conv 64->64, fused relu(BN(.)), bf16 n-major output for b3 GEMM.
// ---------------------------------------------------------------------------
__global__ __launch_bounds__(256)
void conv3x3(const float* __restrict__ X, const float* __restrict__ W,
             const float* __restrict__ bnp, unsigned short* __restrict__ YT)
{
  __shared__ float plane[34 * 34];
  __shared__ float wsh[8 * 9];
  const int tid = threadIdx.x;
  const int bI = blockIdx.x, o0 = blockIdx.y * 8;
  float acc[8][4] = {};
  for (int i = tid; i < 34 * 34; i += 256) plane[i] = 0.f;    // zero halo
  for (int i = 0; i < 64; ++i) {
    __syncthreads();
    for (int e = tid; e < 1024; e += 256) {
      int y = e >> 5, x = e & 31;
      plane[(y + 1) * 34 + x + 1] = X[((size_t)bI * 64 + i) * 1024 + e];
    }
    if (tid < 72) wsh[tid] = W[((size_t)(o0 + tid / 9) * 64 + i) * 9 + tid % 9];
    __syncthreads();
#pragma unroll
    for (int p = 0; p < 4; ++p) {
      int e = tid + 256 * p;
      int y = e >> 5, x = e & 31;
      float t0 = plane[y*34 + x],     t1 = plane[y*34 + x + 1],     t2 = plane[y*34 + x + 2];
      float t3 = plane[(y+1)*34 + x], t4 = plane[(y+1)*34 + x + 1], t5 = plane[(y+1)*34 + x + 2];
      float t6 = plane[(y+2)*34 + x], t7 = plane[(y+2)*34 + x + 1], t8 = plane[(y+2)*34 + x + 2];
#pragma unroll
      for (int oo = 0; oo < 8; ++oo) {
        const float* w = &wsh[oo * 9];
        acc[oo][p] += t0*w[0] + t1*w[1] + t2*w[2] + t3*w[3] + t4*w[4]
                    + t5*w[5] + t6*w[6] + t7*w[7] + t8*w[8];
      }
    }
  }
#pragma unroll
  for (int p = 0; p < 4; ++p) {
    int e = tid + 256 * p;
    unsigned short bt[8];
#pragma unroll
    for (int oo = 0; oo < 8; ++oo) {
      int o = o0 + oo;
      bt[oo] = f2bf1(fmaxf(acc[oo][p] * bnp[o] + bnp[64 + o], 0.f));
    }
    uint4 pk;
    pk.x = (unsigned)bt[0] | ((unsigned)bt[1] << 16);
    pk.y = (unsigned)bt[2] | ((unsigned)bt[3] << 16);
    pk.z = (unsigned)bt[4] | ((unsigned)bt[5] << 16);
    pk.w = (unsigned)bt[6] | ((unsigned)bt[7] << 16);
    *reinterpret_cast<uint4*>(YT + ((size_t)bI * Nc + e) * 64 + o0) = pk;
  }
}

// ---------------------------------------------------------------------------
// Helpers: weight f32 -> bf16, BN param folding (scale/shift)
// ---------------------------------------------------------------------------
__global__ void f2bf_kernel(const float* __restrict__ s, unsigned short* __restrict__ d, int n) {
  int i = blockIdx.x * 256 + threadIdx.x;
  if (i < n) d[i] = f2bf1(s[i]);
}
__global__ void bnprep_kernel(const float* __restrict__ p, float* __restrict__ o, int c) {
  int i = blockIdx.x * 256 + threadIdx.x;
  if (i < c) {
    float g = p[i], bb = p[c + i], m = p[2 * c + i], vv = p[3 * c + i];
    float s = g * rsqrtf(vv + 1e-5f);
    o[i] = s; o[c + i] = bb - m * s;
  }
}

// ---------------------------------------------------------------------------
extern "C" void kernel_launch(void* const* d_in, const int* in_sizes, int n_in,
                              void* d_out, int out_size, void* d_ws, size_t ws_size,
                              hipStream_t stream)
{
  (void)in_sizes; (void)n_in; (void)out_size; (void)ws_size;
  const float* x      = (const float*)d_in[0];
  const float* g1_w   = (const float*)d_in[1];
  const float* g1_bnp = (const float*)d_in[2];
  const float* edge_w = (const float*)d_in[3];
  const float* edge_b = (const float*)d_in[4];
  const float* g2_w   = (const float*)d_in[5];
  const float* g2_bnp = (const float*)d_in[6];
  const float* f1_w   = (const float*)d_in[7];
  const float* f1_b   = (const float*)d_in[8];
  const float* f1_bnp = (const float*)d_in[9];
  const float* f2_w   = (const float*)d_in[10];
  const float* f2_b   = (const float*)d_in[11];
  const float* f2_bnp = (const float*)d_in[12];
  const float* b1_w   = (const float*)d_in[13];
  const float* b1_bnp = (const float*)d_in[14];
  const float* b2_w   = (const float*)d_in[15];
  const float* b2_bnp = (const float*)d_in[16];
  const float* b3_w   = (const float*)d_in[17];
  const float* b3_bnp = (const float*)d_in[18];
  const float* bnf_p  = (const float*)d_in[19];
  float* out = (float*)d_out;

  char* ws = (char*)d_ws;
  size_t off = 0;
  auto alloc = [&](size_t bytes) -> void* {
    void* p = ws + off;
    off = (off + bytes + 255) & ~(size_t)255;
    return p;
  };
  // bf16 weights
  unsigned short* WB_G1 = (unsigned short*)alloc(Cc * Cc * 2);
  unsigned short* WB_E  = (unsigned short*)alloc(512 * 512 * 2);
  unsigned short* WB_G2 = (unsigned short*)alloc(Cc * 512 * 2);
  unsigned short* WB_F1 = (unsigned short*)alloc(1024 * Cc * 2);
  unsigned short* WB_F2 = (unsigned short*)alloc(Cc * 1024 * 2);
  unsigned short* WB_B1 = (unsigned short*)alloc(64 * Cc * 2);
  unsigned short* WB_B3 = (unsigned short*)alloc(Cc * 64 * 2);
  // folded BN params (scale, shift)
  float* BN_G1 = (float*)alloc(2 * 256 * 4);
  float* BN_G2 = (float*)alloc(2 * 256 * 4);
  float* BN_F1 = (float*)alloc(2 * 1024 * 4);
  float* BN_F2 = (float*)alloc(2 * 256 * 4);
  float* BN_B1 = (float*)alloc(2 * 64 * 4);
  float* BN_B2 = (float*)alloc(2 * 64 * 4);
  float* BN_B3 = (float*)alloc(2 * 256 * 4);
  float* BN_F  = (float*)alloc(2 * 256 * 4);
  // activations
  unsigned short* XT   = (unsigned short*)alloc((size_t)NBc * Cc  * 2); // x, bf16 n-major
  unsigned short* FT   = (unsigned short*)alloc((size_t)NBc * Cc  * 2); // g1 out (feat)
  unsigned short* XN   = (unsigned short*)alloc((size_t)NBc * Cc  * 2); // normalized
  int*            IDX  = (int*)           alloc((size_t)NBc * Kk  * 4);
  unsigned short* ET   = (unsigned short*)alloc((size_t)NBc * 512 * 2); // edge out
  float*          H2   = (float*)         alloc((size_t)NBc * Cc  * 4); // g2 out f32
  unsigned short* H2T  = (unsigned short*)alloc((size_t)NBc * Cc  * 2);
  unsigned short* F1T  = (unsigned short*)alloc((size_t)NBc * 1024* 2); // f1 out
  float*          H3   = (float*)         alloc((size_t)NBc * Cc  * 4); // f2 out f32
  unsigned short* H3T  = (unsigned short*)alloc((size_t)NBc * Cc  * 2);
  float*          B1O  = (float*)         alloc((size_t)NBc * 64  * 4);
  unsigned short* B2OT = (unsigned short*)alloc((size_t)NBc * 64  * 2);

  auto cvt = [&](const float* s, unsigned short* d, int n) {
    f2bf_kernel<<<(n + 255) / 256, 256, 0, stream>>>(s, d, n);
  };
  cvt(g1_w, WB_G1, Cc * Cc);
  cvt(edge_w, WB_E, 512 * 512);
  cvt(g2_w, WB_G2, Cc * 512);
  cvt(f1_w, WB_F1, 1024 * Cc);
  cvt(f2_w, WB_F2, Cc * 1024);
  cvt(b1_w, WB_B1, 64 * Cc);
  cvt(b3_w, WB_B3, Cc * 64);

  auto bnp = [&](const float* p, float* o, int c) {
    bnprep_kernel<<<(c + 255) / 256, 256, 0, stream>>>(p, o, c);
  };
  bnp(g1_bnp, BN_G1, 256);  bnp(g2_bnp, BN_G2, 256);
  bnp(f1_bnp, BN_F1, 1024); bnp(f2_bnp, BN_F2, 256);
  bnp(b1_bnp, BN_B1, 64);   bnp(b2_bnp, BN_B2, 64);
  bnp(b3_bnp, BN_B3, 256);  bnp(bnf_p,  BN_F,  256);

  const dim3 blk(256);

  // x -> bf16 n-major
  transpose_bf<<<dim3(Nc / 32, Cc / 32, Bc), blk, 0, stream>>>(x, XT);

  // g1: FT = bf16T( BN(g1_w @ x) )
  gemm_bt<false, false, true, false, false, false, false, true>
      <<<dim3(NBc / 256, Cc / 32), blk, 0, stream>>>(
          (const unsigned*)WB_G1, (const unsigned*)XT, nullptr, FT,
          nullptr, BN_G1, nullptr, nullptr, nullptr, Cc, Cc);

  // normalized features, similarity strips + top-9, edge MLP
  rownorm<<<(NBc) / 8, blk, 0, stream>>>(FT, XN);
  sim_topk<<<dim3(Nc / 16, Bc), blk, 0, stream>>>((const unsigned*)XN, IDX);
  edge_gemm<<<dim3(Bc * (Nc / 16)), blk, 0, stream>>>(
      (const unsigned*)FT, IDX, (const unsigned*)WB_E, edge_b, ET);

  // g2: h2 = BN(g2_w @ e) + x   (f32 + bf16T)
  gemm_bt<false, false, true, true, false, false, true, true>
      <<<dim3(NBc / 256, Cc / 32), blk, 0, stream>>>(
          (const unsigned*)WB_G2, (const unsigned*)ET, H2, H2T,
          nullptr, BN_G2, x, nullptr, nullptr, 512, Cc);

  // f1: relu(BN(f1_w @ h2 + f1_b))   (bf16T only)
  gemm_bt<true, true, true, false, false, false, false, true>
      <<<dim3(NBc / 256, 1024 / 32), blk, 0, stream>>>(
          (const unsigned*)WB_F1, (const unsigned*)H2T, nullptr, F1T,
          f1_b, BN_F1, nullptr, nullptr, nullptr, Cc, 1024);

  // f2: h3 = BN(f2_w @ f1o + f2_b) + h2   (f32 + bf16T)
  gemm_bt<false, true, true, true, false, false, true, true>
      <<<dim3(NBc / 256, Cc / 32), blk, 0, stream>>>(
          (const unsigned*)WB_F2, (const unsigned*)F1T, H3, H3T,
          f2_b, BN_F2, H2, nullptr, nullptr, 1024, Cc);

  // b1: relu(BN(b1_w @ h3))   (f32 only, feeds conv)
  gemm_bt<true, false, true, false, false, false, true, false>
      <<<dim3(NBc / 256, 64 / 32), blk, 0, stream>>>(
          (const unsigned*)WB_B1, (const unsigned*)H3T, B1O, nullptr,
          nullptr, BN_B1, nullptr, nullptr, nullptr, Cc, 64);

  // b2: relu(BN(conv3x3(b1o)))  (bf16T for b3)
  conv3x3<<<dim3(Bc, 8), blk, 0, stream>>>(B1O, b2_w, BN_B2, B2OT);

  // b3 + final: out = BNf( BN3(b3_w @ b2o) + h3 + x )
  gemm_bt<false, false, true, true, true, true, true, false>
      <<<dim3(NBc / 256, Cc / 32), blk, 0, stream>>>(
          (const unsigned*)WB_B3, (const unsigned*)B2OT, out, nullptr,
          nullptr, BN_B3, H3, x, BN_F, 64, Cc);
}
